// StepLayer_80934363726108
// MI455X (gfx1250) — matrix operand order repaired and checked by reading, not compiled, not run on hardware
//
#include <hip/hip_runtime.h>
#include <hip/hip_bf16.h>

// ---------------------------------------------------------------------------
// CDNA5 (gfx1250) transformer encoder layer, bf16 WMMA with f32 accumulate.
// ---------------------------------------------------------------------------

typedef __attribute__((ext_vector_type(16))) __bf16 v16bf;
typedef __attribute__((ext_vector_type(8)))  float  v8f;

union Frag16 { v16bf v; unsigned int u[8]; unsigned short h[16]; };

// Async global->LDS (CDNA5 GLOBAL_LOAD_ASYNC_TO_LDS_B128, ASYNCcnt-tracked).
// Probe round 2 confirmed the builtin exists and expects v4i pointers:
//   (v4i addrspace(1)* src, v4i addrspace(3)* dst, imm offset, imm cpol)
#if defined(__has_builtin)
#if __has_builtin(__builtin_amdgcn_global_load_async_to_lds_b128) && \
    __has_builtin(__builtin_amdgcn_s_wait_asynccnt)
#define USE_ASYNC_LDS 1
#endif
#endif
#ifndef USE_ASYNC_LDS
#define USE_ASYNC_LDS 0
#endif

#if USE_ASYNC_LDS
typedef __attribute__((__vector_size__(4 * sizeof(int)))) int vi4;
typedef __attribute__((address_space(1))) vi4 as1_vi4;
typedef __attribute__((address_space(3))) vi4 as3_vi4;
#endif

__device__ __forceinline__ unsigned short f32_bf16(float f) {
  union { float f; unsigned int u; } x; x.f = f;
  unsigned int u = x.u;
  unsigned int r = u + 0x7FFFu + ((u >> 16) & 1u);   // round-to-nearest-even
  return (unsigned short)(r >> 16);
}

// A-matrix (16x32 bf16) K index for VGPR v, lane-half hf  (ISA 7.12.2)
__device__ __forceinline__ int kbA(int v, int hf) {
  return (v < 4 ? 2 * v : 16 + 2 * (v - 4)) + 8 * hf;
}
// B-matrix (32x16 bf16) K index for VGPR v, lane-half hf
__device__ __forceinline__ int kbB(int v, int hf) {
  return (v < 4 ? 2 * v : 16 + 2 * (v - 4)) + 8 * hf;
}

__device__ __forceinline__ v8f wmma_bf16(v16bf a, v16bf b, v8f c) {
  return __builtin_amdgcn_wmma_f32_16x16x32_bf16(false, a, false, b,
                                                 (short)0, c, false, false);
}

// ---------------------------------------------------------------------------
// fp32 -> bf16 elementwise convert (weights)
// ---------------------------------------------------------------------------
__global__ void cvt_kernel(const float* __restrict__ in,
                           unsigned short* __restrict__ out, int n) {
  int i = blockIdx.x * blockDim.x + threadIdx.x;
  if (i < n) out[i] = f32_bf16(in[i]);
}

// ---------------------------------------------------------------------------
// LayerNorm (optionally fused residual add), row length 1024, out bf16
// ---------------------------------------------------------------------------
__global__ __launch_bounds__(256) void ln_kernel(
    const float* __restrict__ in, const float* __restrict__ res,
    const float* __restrict__ g, const float* __restrict__ bb,
    unsigned short* __restrict__ out) {
  const int d = 1024;
  const int row = blockIdx.x, t = threadIdx.x;
  const float* ip = in + (size_t)row * d;
  float v0[4];
#pragma unroll
  for (int i = 0; i < 4; i++) {
    int j = t + i * 256;
    float a = ip[j];
    if (res) a += res[(size_t)row * d + j];
    v0[i] = a;
  }
  __shared__ float red[256];
  red[t] = v0[0] + v0[1] + v0[2] + v0[3];
  __syncthreads();
  for (int o = 128; o > 0; o >>= 1) {
    if (t < o) red[t] += red[t + o];
    __syncthreads();
  }
  float mu = red[0] * (1.0f / d);
  __syncthreads();
  float q = 0.f;
#pragma unroll
  for (int i = 0; i < 4; i++) { float dv = v0[i] - mu; q += dv * dv; }
  red[t] = q;
  __syncthreads();
  for (int o = 128; o > 0; o >>= 1) {
    if (t < o) red[t] += red[t + o];
    __syncthreads();
  }
  float rs = rsqrtf(red[0] * (1.0f / d) + 1e-5f);
#pragma unroll
  for (int i = 0; i < 4; i++) {
    int j = t + i * 256;
    float y = (v0[i] - mu) * rs * g[j] + bb[j];
    out[(size_t)row * d + j] = f32_bf16(y);
  }
}

// ---------------------------------------------------------------------------
// bf16 WMMA GEMM, C = A(MxK) * B(KxN) + bias, fused epilogues.
//   mode 0: bf16 head-split (B,H,S,dk)            (Q / K projections)
//   mode 1: bf16 head-split transposed (B,H,dk,S) (V projection)
//   mode 2: f32 row-major                         (attn_out = ctx @ Wo + bo)
//   mode 3: bf16 row-major, exact GELU            (FFN hidden)
//   mode 4: f32 row-major, + res1 + res2          (final output)
// Block: 256 threads (8 waves), tile 256(M) x 64(N), K step 32.
// Each wave owns a 32x64 strip: 2 A-frags x 4 B-frags -> 8 WMMAs / K-step.
// ---------------------------------------------------------------------------
#define BM 256
#define BN 64
#define BK 32

__global__ __launch_bounds__(256) void gemm_kernel(
    const unsigned short* __restrict__ A, const unsigned short* __restrict__ Bw,
    const float* __restrict__ bias, int M, int N, int K, int mode,
    float* __restrict__ outF, unsigned short* __restrict__ outBF,
    const float* __restrict__ res1, const float* __restrict__ res2,
    int S, int H, int dk) {
  __shared__ alignas(16) unsigned short lA[BM * BK];   // [row][k]   16 KB
  __shared__ alignas(16) unsigned short lBt[BN * BK];  // [n][k]      4 KB
  const int t = threadIdx.x;
  const int wv = t >> 5, lane = t & 31, m16 = lane & 15, hf = lane >> 4;
  const int bM = blockIdx.x * BM, bN = blockIdx.y * BN;

  v8f acc[2][4] = {};

  for (int k0 = 0; k0 < K; k0 += BK) {
    // A tile: 256x32 bf16 = 16 KB, 4 x 16-byte chunks per thread
#pragma unroll
    for (int i = 0; i < 4; i++) {
      int e = (t + i * 256) * 8;
      int r = e >> 5, c = e & 31;
      const unsigned short* gp = &A[(size_t)(bM + r) * K + k0 + c];
      unsigned short* lp = &lA[r * 32 + c];
#if USE_ASYNC_LDS
      __builtin_amdgcn_global_load_async_to_lds_b128(
          (as1_vi4*)gp, (as3_vi4*)lp, 0, 0);
#else
      *(uint4*)lp = *(const uint4*)gp;
#endif
    }
    // B tile: 32x64 bf16, transpose into lBt[n][k]
    {
      int e = t * 8;
      int r = e >> 6, c = e & 63;
      uint4 dat = *(const uint4*)(&Bw[(size_t)(k0 + r) * N + bN + c]);
      const unsigned short* ds = (const unsigned short*)&dat;
#pragma unroll
      for (int j = 0; j < 8; j++) lBt[(c + j) * 32 + r] = ds[j];
    }
#if USE_ASYNC_LDS
    __builtin_amdgcn_s_wait_asynccnt(0);
#endif
    __syncthreads();

    Frag16 bfr[4];
#pragma unroll
    for (int nt = 0; nt < 4; nt++)
#pragma unroll
      for (int v = 0; v < 8; v++)
        bfr[nt].u[v] =
            *(const unsigned int*)(&lBt[(nt * 16 + m16) * 32 + kbB(v, hf)]);
#pragma unroll
    for (int mt = 0; mt < 2; mt++) {
      Frag16 af;
#pragma unroll
      for (int v = 0; v < 8; v++)
        af.u[v] = *(const unsigned int*)(
            &lA[(wv * 32 + mt * 16 + m16) * 32 + kbA(v, hf)]);
#pragma unroll
      for (int nt = 0; nt < 4; nt++)
        acc[mt][nt] = wmma_bf16(af.v, bfr[nt].v, acc[mt][nt]);
    }
    __syncthreads();
  }

  // epilogue
#pragma unroll
  for (int mt = 0; mt < 2; mt++) {
#pragma unroll
    for (int nt = 0; nt < 4; nt++) {
      int col = bN + nt * 16 + m16;
      float bv = bias[col];
#pragma unroll
      for (int v = 0; v < 8; v++) {
        int row = bM + wv * 32 + mt * 16 + v + 8 * hf;
        float val = acc[mt][nt][v] + bv;
        if (mode == 0 || mode == 1) {
          int bidx = row / S, sidx = row % S;
          int hh = col / dk, off = col % dk;
          size_t idx =
              (mode == 0)
                  ? (((size_t)(bidx * H + hh) * S + sidx) * dk + off)
                  : (((size_t)(bidx * H + hh) * dk + off) * S + sidx);
          outBF[idx] = f32_bf16(val);
        } else if (mode == 2) {
          outF[(size_t)row * N + col] = val;
        } else if (mode == 3) {
          float gel = 0.5f * val * (1.0f + erff(val * 0.70710678118f));
          outBF[(size_t)row * N + col] = f32_bf16(gel);
        } else {
          size_t idx = (size_t)row * N + col;
          outF[idx] = res1[idx] + res2[idx] + val;
        }
      }
    }
  }
}

// ---------------------------------------------------------------------------
// Flash attention: one wave per 16-query tile, keys streamed 32 at a time.
// q,k: (B,H,S,64) bf16 ; vT: (B,H,64,S) bf16 ; mask: (B,S) int ;
// ctx out: (B,S,1024) bf16
// ---------------------------------------------------------------------------
__global__ __launch_bounds__(128) void attn_kernel(
    const unsigned short* __restrict__ q, const unsigned short* __restrict__ k,
    const unsigned short* __restrict__ vT, const int* __restrict__ mask,
    unsigned short* __restrict__ ctx, int B, int H, int S, int dmodel) {
  const int dk = 64;
  __shared__ alignas(16) unsigned short pbuf[4][16 * 32];
  const int wv = threadIdx.x >> 5, lane = threadIdx.x & 31;
  const int m16 = lane & 15, hf = lane >> 4;
  const int wid = blockIdx.x * 4 + wv;
  const int nqt = S / 16;
  const int qt = wid % nqt;
  const int hh = (wid / nqt) % H;
  const int bb = wid / (nqt * H);
  const int qbase = qt * 16;
  const size_t bh = (size_t)(bb * H + hh);
  const unsigned short* qp = q + bh * S * dk;
  const unsigned short* kp = k + bh * S * dk;
  const unsigned short* vp = vT + bh * dk * S;
  const int* mp = mask + bb * S;
  unsigned short* pb = pbuf[wv];

  // Q fragments (16 x 64, split as two 16x32 A-frags), reused for all keys
  Frag16 qf[2];
#pragma unroll
  for (int kk = 0; kk < 2; kk++)
#pragma unroll
    for (int v = 0; v < 8; v++)
      qf[kk].u[v] = *(const unsigned int*)(
          &qp[(size_t)(qbase + m16) * dk + kk * 32 + kbA(v, hf)]);

  v8f o[4] = {};
  float mrow[8], lrow[8];
#pragma unroll
  for (int v = 0; v < 8; v++) { mrow[v] = -3.0e38f; lrow[v] = 0.f; }
  const float scale = 0.125f;  // 1/sqrt(64)

  for (int c0 = 0; c0 < S; c0 += 32) {
    v8f s0 = {}, s1 = {};
#pragma unroll
    for (int kk = 0; kk < 2; kk++) {
      Frag16 kf0, kf1;
#pragma unroll
      for (int v = 0; v < 8; v++) {
        kf0.u[v] = *(const unsigned int*)(
            &kp[(size_t)(c0 + m16) * dk + kk * 32 + kbB(v, hf)]);
        kf1.u[v] = *(const unsigned int*)(
            &kp[(size_t)(c0 + 16 + m16) * dk + kk * 32 + kbB(v, hf)]);
      }
      s0 = wmma_bf16(qf[kk].v, kf0.v, s0);
      s1 = wmma_bf16(qf[kk].v, kf1.v, s1);
    }

    const int mk0 = mp[c0 + m16];
    const int mk1 = mp[c0 + 16 + m16];
#pragma unroll
    for (int v = 0; v < 8; v++) {
      float a0 = mk0 ? s0[v] * scale : -1.0e9f;
      float a1 = mk1 ? s1[v] * scale : -1.0e9f;
      float mx = fmaxf(a0, a1);
#pragma unroll
      for (int off = 1; off < 16; off <<= 1)
        mx = fmaxf(mx, __shfl_xor(mx, off, 32));
      float mnew = fmaxf(mrow[v], mx);
      float cr = __expf(mrow[v] - mnew);
      float p0 = __expf(a0 - mnew);
      float p1 = __expf(a1 - mnew);
      float rs = p0 + p1;
#pragma unroll
      for (int off = 1; off < 16; off <<= 1) rs += __shfl_xor(rs, off, 32);
      lrow[v] = lrow[v] * cr + rs;
      mrow[v] = mnew;
      int r = v + 8 * hf;                 // C-layout row -> LDS row
      pb[r * 32 + m16] = f32_bf16(p0);
      pb[r * 32 + 16 + m16] = f32_bf16(p1);
      o[0][v] *= cr; o[1][v] *= cr; o[2][v] *= cr; o[3][v] *= cr;
    }

    // relayout P (C-layout) -> A-fragment via per-wave LDS (same-wave DS
    // ops are in order: no barrier needed)
    Frag16 pf;
#pragma unroll
    for (int v = 0; v < 8; v++)
      pf.u[v] = *(const unsigned int*)(&pb[m16 * 32 + kbA(v, hf)]);

#pragma unroll
    for (int nt = 0; nt < 4; nt++) {
      Frag16 vf;
#pragma unroll
      for (int v = 0; v < 8; v++)
        vf.u[v] = *(const unsigned int*)(
            &vp[(size_t)(nt * 16 + m16) * S + c0 + kbB(v, hf)]);
      o[nt] = wmma_bf16(pf.v, vf.v, o[nt]);
    }
  }

#pragma unroll
  for (int v = 0; v < 8; v++) {
    float inv = 1.0f / lrow[v];
    int r = qbase + v + 8 * hf;
#pragma unroll
    for (int nt = 0; nt < 4; nt++)
      ctx[((size_t)(bb * S + r)) * dmodel + hh * 64 + nt * 16 + m16] =
          f32_bf16(o[nt][v] * inv);
  }
}

// ---------------------------------------------------------------------------
// Host-side orchestration
// ---------------------------------------------------------------------------
extern "C" void kernel_launch(void* const* d_in, const int* in_sizes, int n_in,
                              void* d_out, int out_size, void* d_ws,
                              size_t ws_size, hipStream_t stream) {
  (void)in_sizes; (void)n_in; (void)out_size; (void)ws_size;
  const int B = 4, S = 2048, D = 1024, H = 16, DK = 64, F = 4096;
  const int M = B * S;  // 8192 token rows

  const float* x     = (const float*)d_in[0];
  const int*   mask  = (const int*)d_in[1];
  const float* Wq    = (const float*)d_in[2];
  const float* bq    = (const float*)d_in[3];
  const float* Wk    = (const float*)d_in[4];
  const float* bk    = (const float*)d_in[5];
  const float* Wv    = (const float*)d_in[6];
  const float* bv    = (const float*)d_in[7];
  const float* Wo    = (const float*)d_in[8];
  const float* bo    = (const float*)d_in[9];
  const float* ln1g  = (const float*)d_in[10];
  const float* ln1b  = (const float*)d_in[11];
  const float* ln2g  = (const float*)d_in[12];
  const float* ln2b  = (const float*)d_in[13];
  const float* W1    = (const float*)d_in[14];
  const float* b1    = (const float*)d_in[15];
  const float* W2    = (const float*)d_in[16];
  const float* b2    = (const float*)d_in[17];
  float* out = (float*)d_out;

  size_t off = 0;
  char* base = (char*)d_ws;
  auto alloc = [&](size_t bytes) -> void* {
    void* p = base + off;
    off += (bytes + 255) & ~(size_t)255;
    return p;
  };
  unsigned short* x1b = (unsigned short*)alloc((size_t)M * D * 2);
  unsigned short* Wqb = (unsigned short*)alloc((size_t)D * D * 2);
  unsigned short* Wkb = (unsigned short*)alloc((size_t)D * D * 2);
  unsigned short* Wvb = (unsigned short*)alloc((size_t)D * D * 2);
  unsigned short* Wob = (unsigned short*)alloc((size_t)D * D * 2);
  unsigned short* W1b = (unsigned short*)alloc((size_t)D * F * 2);
  unsigned short* W2b = (unsigned short*)alloc((size_t)F * D * 2);
  unsigned short* qb  = (unsigned short*)alloc((size_t)M * D * 2);
  unsigned short* kbf = (unsigned short*)alloc((size_t)M * D * 2);
  unsigned short* vTb = (unsigned short*)alloc((size_t)M * D * 2);
  unsigned short* ctxb = (unsigned short*)alloc((size_t)M * D * 2);
  float* attn = (float*)alloc((size_t)M * D * 4);
  unsigned short* x2b = (unsigned short*)alloc((size_t)M * D * 2);
  unsigned short* hb  = (unsigned short*)alloc((size_t)M * F * 2);

  // 1. weights -> bf16
  const int ndd = D * D, ndf = D * F;
  cvt_kernel<<<(ndd + 255) / 256, 256, 0, stream>>>(Wq, Wqb, ndd);
  cvt_kernel<<<(ndd + 255) / 256, 256, 0, stream>>>(Wk, Wkb, ndd);
  cvt_kernel<<<(ndd + 255) / 256, 256, 0, stream>>>(Wv, Wvb, ndd);
  cvt_kernel<<<(ndd + 255) / 256, 256, 0, stream>>>(Wo, Wob, ndd);
  cvt_kernel<<<(ndf + 255) / 256, 256, 0, stream>>>(W1, W1b, ndf);
  cvt_kernel<<<(ndf + 255) / 256, 256, 0, stream>>>(W2, W2b, ndf);

  // 2. LN1 -> bf16
  ln_kernel<<<M, 256, 0, stream>>>(x, nullptr, ln1g, ln1b, x1b);

  // 3. QKV projections
  dim3 gD(M / BM, D / BN);
  gemm_kernel<<<gD, 256, 0, stream>>>(x1b, Wqb, bq, M, D, D, 0, nullptr, qb,
                                      nullptr, nullptr, S, H, DK);
  gemm_kernel<<<gD, 256, 0, stream>>>(x1b, Wkb, bk, M, D, D, 0, nullptr, kbf,
                                      nullptr, nullptr, S, H, DK);
  gemm_kernel<<<gD, 256, 0, stream>>>(x1b, Wvb, bv, M, D, D, 1, nullptr, vTb,
                                      nullptr, nullptr, S, H, DK);

  // 4. flash attention -> ctx (B,S,D) bf16
  int nwaves = B * H * (S / 16);
  attn_kernel<<<nwaves / 4, 128, 0, stream>>>(qb, kbf, vTb, mask, ctxb, B, H,
                                              S, D);

  // 5. output projection -> attn_out f32
  gemm_kernel<<<gD, 256, 0, stream>>>(ctxb, Wob, bo, M, D, D, 2, attn, nullptr,
                                      nullptr, nullptr, S, H, DK);

  // 6. LN2(x + attn_out) -> bf16
  ln_kernel<<<M, 256, 0, stream>>>(x, attn, ln2g, ln2b, x2b);

  // 7. FFN up + GELU -> bf16
  dim3 gF(M / BM, F / BN);
  gemm_kernel<<<gF, 256, 0, stream>>>(x2b, W1b, b1, M, F, D, 3, nullptr, hb,
                                      nullptr, nullptr, S, H, DK);

  // 8. FFN down + x + attn_out -> out f32
  gemm_kernel<<<gD, 256, 0, stream>>>(hb, W2b, b2, M, D, F, 4, out, nullptr,
                                      x, attn, S, H, DK);
}
